// MultiHeadSelfAttention_88596585382709
// MI455X (gfx1250) — matrix operand order repaired and checked by reading, not compiled, not run on hardware
//
#include <hip/hip_runtime.h>

#define DMODEL 2048
#define NHEADS 32
#define NKV    8
#define HDIM   64
#define BATCH  2
#define SEQ    2048
#define MROWS  (BATCH * SEQ)   // 4096
#define KVDIM  (NKV * HDIM)    // 512

typedef __attribute__((ext_vector_type(16))) __bf16 v16bf;
typedef __attribute__((ext_vector_type(8)))  float  v8f;

__device__ __forceinline__ unsigned short f2bf(float f) {
  unsigned u = __float_as_uint(f);
  u += 0x7fffu + ((u >> 16) & 1u);          // round-to-nearest-even
  return (unsigned short)(u >> 16);
}
__device__ __forceinline__ float bf2f(unsigned short h) {
  return __uint_as_float(((unsigned)h) << 16);
}

// gfx1250 async global->LDS copy (ASYNCcnt), 16 bytes per lane.
// Low 32 bits of a flat LDS address are the LDS offset (aperture check uses
// addr[63:32] only), so truncation gives the DS-relative address for VDST.
__device__ __forceinline__ void async_cp16(void* lds_dst, const void* gsrc) {
  unsigned ldsa = (unsigned)(size_t)lds_dst;
  asm volatile("global_load_async_to_lds_b128 %0, %1, off"
               :: "v"(ldsa), "v"(gsrc)
               : "memory");
}
__device__ __forceinline__ void async_wait0() {
  asm volatile("s_wait_asynccnt 0x0" ::: "memory");
}

// ---------------------------------------------------------------- conversions
__global__ void cvt_f32_bf16(const float* __restrict__ in,
                             unsigned short* __restrict__ out, int n) {
  for (int i = blockIdx.x * blockDim.x + threadIdx.x; i < n;
       i += gridDim.x * blockDim.x)
    out[i] = f2bf(in[i]);
}

// ------------------------------------------------------------------- GEMM NT
// C[M,N] = A[M,K] * W[N,K]^T  (A, W bf16 row-major, K-contiguous both sides)
// Block: 128x128, 256 threads (8 waves, 4x2), wave tile 32x64, K-step 32.
// A tile: async copy to LDS [m][k].   B tile: register transpose to LDS [k][n].
template <bool F32OUT>
__global__ __launch_bounds__(256) void gemm_bf16_nt(
    const unsigned short* __restrict__ A,
    const unsigned short* __restrict__ W,
    void* __restrict__ C, int M, int N, int K) {
  __shared__ __align__(16) unsigned short As[128][32];       // [m][k]
  __shared__ __align__(16) unsigned short Bs[32][136];       // [k][n], pad 8

  const int tid  = threadIdx.x;
  const int lane = tid & 31;
  const int wave = tid >> 5;
  const int wm   = wave >> 1;           // 0..3
  const int wn   = wave & 1;            // 0..1
  const int half = lane >> 4;
  const int lr   = lane & 15;
  const int m0   = blockIdx.y * 128;
  const int n0   = blockIdx.x * 128;
  const int nk   = K >> 5;

  v8f acc[2][4];
#pragma unroll
  for (int mt = 0; mt < 2; ++mt)
#pragma unroll
    for (int nt = 0; nt < 4; ++nt)
      acc[mt][nt] = (v8f){0.f, 0.f, 0.f, 0.f, 0.f, 0.f, 0.f, 0.f};

  union AU { v16bf v; uint4 q[2]; };
  union BU { v16bf v; uint4 q[2]; };

  for (int kt = 0; kt < nk; ++kt) {
    const int kk0 = kt << 5;
    // A tile: async global->LDS, 16B chunks. chunk c = tid + 256*i
#pragma unroll
    for (int i = 0; i < 2; ++i) {
      int c = tid + (i << 8);
      int m = c >> 2, k0 = (c & 3) << 3;
      async_cp16(&As[m][k0], &A[(size_t)(m0 + m) * K + kk0 + k0]);
    }
    // B tile: load b128, transpose-scatter into Bs[k][n]
#pragma unroll
    for (int i = 0; i < 2; ++i) {
      int c = tid + (i << 8);
      int n = c >> 2, k0 = (c & 3) << 3;
      union { uint4 q; unsigned short s[8]; } w;
      w.q = *(const uint4*)&W[(size_t)(n0 + n) * K + kk0 + k0];
#pragma unroll
      for (int j = 0; j < 8; ++j) Bs[k0 + j][n] = w.s[j];
      if (kk0 + 32 < K) {   // prefetch next K tile
        __builtin_prefetch(&A[(size_t)(m0 + n) * K + kk0 + 32 + k0], 0, 0);
        __builtin_prefetch(&W[(size_t)(n0 + n) * K + kk0 + 32 + k0], 0, 0);
      }
    }
    async_wait0();
    __syncthreads();

    // A fragments: lane holds row M=lane%16; pairs = two contiguous 8-elt runs
    AU af[2];
#pragma unroll
    for (int mt = 0; mt < 2; ++mt) {
      const unsigned short* ap = &As[wm * 32 + mt * 16 + lr][0];
      af[mt].q[0] = *(const uint4*)(ap + 8 * half);
      af[mt].q[1] = *(const uint4*)(ap + 16 + 8 * half);
    }
    // B fragments: lane holds N=lane%16; K in table order (8h+s, 16+8h+s-8)
    BU bfv[4];
#pragma unroll
    for (int nt = 0; nt < 4; ++nt) {
#pragma unroll
      for (int s = 0; s < 16; ++s)
        ((unsigned short*)&bfv[nt])[s] =
            Bs[s < 8 ? 8 * half + s : 16 + 8 * half + (s - 8)][wn * 64 + nt * 16 + lr];
    }

#pragma unroll
    for (int mt = 0; mt < 2; ++mt)
#pragma unroll
      for (int nt = 0; nt < 4; ++nt)
        acc[mt][nt] = __builtin_amdgcn_wmma_f32_16x16x32_bf16(
            false, af[mt].v, false, bfv[nt].v, (short)0, acc[mt][nt], false,
            false);

    __syncthreads();
  }

  // C layout: VGPR r, lane -> M = r + 8*(lane>=16), N = lane%16
#pragma unroll
  for (int mt = 0; mt < 2; ++mt)
#pragma unroll
    for (int nt = 0; nt < 4; ++nt)
#pragma unroll
      for (int r = 0; r < 8; ++r) {
        int row = m0 + wm * 32 + mt * 16 + r + 8 * half;
        int col = n0 + wn * 64 + nt * 16 + lr;
        if (F32OUT)
          ((float*)C)[(size_t)row * N + col] = acc[mt][nt][r];
        else
          ((unsigned short*)C)[(size_t)row * N + col] = f2bf(acc[mt][nt][r]);
      }
}

// -------------------------------------------------- RoPE + head-major scatter
__global__ void rope_scatter(const unsigned short* __restrict__ in,
                             unsigned short* __restrict__ out, int nh,
                             int do_rope) {
  const int ncols = nh * HDIM;
  const int prow  = ncols >> 1;
  const int total = MROWS * prow;
  for (int p = blockIdx.x * blockDim.x + threadIdx.x; p < total;
       p += gridDim.x * blockDim.x) {
    int row = p / prow;
    int pc  = p % prow;
    int h   = pc / (HDIM / 2);
    int j   = pc % (HDIM / 2);
    int b = row / SEQ, s = row % SEQ;
    float e = bf2f(in[(size_t)row * ncols + h * HDIM + 2 * j]);
    float o = bf2f(in[(size_t)row * ncols + h * HDIM + 2 * j + 1]);
    float oe, oo;
    if (do_rope) {
      float inv = exp2f(-13.287712379549449f * (float)(2 * j) * (1.0f / HDIM));
      float ang = (float)s * inv;
      float c = cosf(ang), sn = sinf(ang);
      oe = e * c - o * sn;
      oo = e * sn + o * c;
    } else {
      oe = e;
      oo = o;
    }
    size_t ob = (((size_t)b * nh + h) * SEQ + s) * HDIM + 2 * j;
    out[ob]     = f2bf(oe);
    out[ob + 1] = f2bf(oo);
  }
}

// ------------------------------------------------------------ flash attention
// Q: [B][H][S][64]  K,V: [B][KV][S][64]  O: [B][S][H*64]  (all bf16)
// 128 threads = 4 waves; block owns 64 query rows (16/wave); 32-key LDS tiles.
__global__ __launch_bounds__(128) void flash_attn(
    const unsigned short* __restrict__ Q, const unsigned short* __restrict__ Kc,
    const unsigned short* __restrict__ Vc, unsigned short* __restrict__ O) {
  __shared__ __align__(16) unsigned short Kt[64][40];   // [dh][key], pad 8
  __shared__ __align__(16) unsigned short Vt[32][72];   // [key][dh], pad 8
  __shared__ __align__(16) unsigned short Pws[4][16][32];

  const int tid  = threadIdx.x;
  const int lane = tid & 31;
  const int wave = tid >> 5;
  const int half = lane >> 4;
  const int lr   = lane & 15;
  const int h    = blockIdx.y;
  const int b    = blockIdx.z;
  const int kvh  = h >> 2;  // groups = 4
  const int qbase = blockIdx.x * 64 + wave * 16;
  const size_t qoff  = ((size_t)b * NHEADS + h) * SEQ * HDIM;
  const size_t kvoff = ((size_t)b * NKV + kvh) * SEQ * HDIM;

  union AU { v16bf v; uint4 q[2]; };
  union BU { v16bf v; uint4 q[2]; };

  // Q A-fragments (K-dim = 64 -> two 32-chunks), two b128 loads each
  AU qa[2];
#pragma unroll
  for (int j = 0; j < 2; ++j) {
    const unsigned short* qp = &Q[qoff + (size_t)(qbase + lr) * HDIM + j * 32];
    qa[j].q[0] = *(const uint4*)(qp + 8 * half);
    qa[j].q[1] = *(const uint4*)(qp + 16 + 8 * half);
  }

  float mrun[8], lrun[8];
  v8f oacc[4];
#pragma unroll
  for (int r = 0; r < 8; ++r) { mrun[r] = -3.0e38f; lrun[r] = 0.f; }
#pragma unroll
  for (int nt = 0; nt < 4; ++nt)
    oacc[nt] = (v8f){0.f, 0.f, 0.f, 0.f, 0.f, 0.f, 0.f, 0.f};

  const float scale = 0.125f;               // 1/sqrt(64)
  const float L2E   = 1.4426950408889634f;
  const int ntiles  = blockIdx.x * 2 + 2;   // block-uniform causal trip count

  for (int kt = 0; kt < ntiles; ++kt) {
    const int key0 = kt * 32;
    // ---- stage K (transposed) and V (async) tiles: 32 keys x 64 dh
#pragma unroll
    for (int i = 0; i < 2; ++i) {
      int c = tid + (i << 7);
      int key = c >> 3, dh0 = (c & 7) << 3;
      union { uint4 q; unsigned short s[8]; } kr;
      kr.q = *(const uint4*)&Kc[kvoff + (size_t)(key0 + key) * HDIM + dh0];
#pragma unroll
      for (int j = 0; j < 8; ++j) Kt[dh0 + j][key] = kr.s[j];
      async_cp16(&Vt[key][dh0],
                 &Vc[kvoff + (size_t)(key0 + key) * HDIM + dh0]);
    }
    async_wait0();
    __syncthreads();

    if (key0 <= qbase + 15) {  // tile intersects this wave's causal region
      v8f s0 = (v8f){0.f, 0.f, 0.f, 0.f, 0.f, 0.f, 0.f, 0.f};
      v8f s1 = s0;
      // scores: S = Q * K^T (contraction over HDIM); B-frag lane = key, dh in table order
#pragma unroll
      for (int j = 0; j < 2; ++j) {
        BU kb0, kb1;
#pragma unroll
        for (int s = 0; s < 16; ++s) {
          int kd = j * 32 + (s < 8 ? 8 * half + s : 16 + 8 * half + (s - 8));
          ((unsigned short*)&kb0)[s] = Kt[kd][lr];
          ((unsigned short*)&kb1)[s] = Kt[kd][16 + lr];
        }
        s0 = __builtin_amdgcn_wmma_f32_16x16x32_bf16(
            false, qa[j].v, false, kb0.v, (short)0, s0, false, false);
        s1 = __builtin_amdgcn_wmma_f32_16x16x32_bf16(
            false, qa[j].v, false, kb1.v, (short)0, s1, false, false);
      }
      // online softmax per row (row r+8*half lives in this lane's VGPR r)
#pragma unroll
      for (int r = 0; r < 8; ++r) {
        int qrow = qbase + r + 8 * half;
        float a = s0[r] * scale;
        float c = s1[r] * scale;
        if (key0 + lr > qrow)      a = -3.0e38f;
        if (key0 + 16 + lr > qrow) c = -3.0e38f;
        float rm = fmaxf(a, c);
#pragma unroll
        for (int m = 1; m <= 8; m <<= 1) rm = fmaxf(rm, __shfl_xor(rm, m));
        float newm  = fmaxf(mrun[r], rm);
        float alpha = exp2f((mrun[r] - newm) * L2E);
        float p0 = exp2f((a - newm) * L2E);
        float p1 = exp2f((c - newm) * L2E);
        float rs = p0 + p1;
#pragma unroll
        for (int m = 1; m <= 8; m <<= 1) rs += __shfl_xor(rs, m);
        lrun[r] = lrun[r] * alpha + rs;
        mrun[r] = newm;
#pragma unroll
        for (int nt = 0; nt < 4; ++nt) oacc[nt][r] *= alpha;
        Pws[wave][r + 8 * half][lr]      = f2bf(p0);
        Pws[wave][r + 8 * half][16 + lr] = f2bf(p1);
      }
      // P back as A-fragment (wave-private LDS region, DS in-order per wave)
      AU pa;
      {
        const unsigned short* pp = &Pws[wave][lr][0];
        pa.q[0] = *(const uint4*)(pp + 8 * half);
        pa.q[1] = *(const uint4*)(pp + 16 + 8 * half);
      }
      // O += P * V  (contraction over 32 keys; lane = dh, keys in table order)
#pragma unroll
      for (int nt = 0; nt < 4; ++nt) {
        BU vb;
#pragma unroll
        for (int s = 0; s < 16; ++s)
          ((unsigned short*)&vb)[s] =
              Vt[s < 8 ? 8 * half + s : 16 + 8 * half + (s - 8)][nt * 16 + lr];
        oacc[nt] = __builtin_amdgcn_wmma_f32_16x16x32_bf16(
            false, pa.v, false, vb.v, (short)0, oacc[nt], false, false);
      }
    }
    __syncthreads();
  }

  // epilogue: normalize and scatter to [B][S][H*64]
#pragma unroll
  for (int nt = 0; nt < 4; ++nt)
#pragma unroll
    for (int r = 0; r < 8; ++r) {
      int srow = qbase + r + 8 * half;
      int col  = h * HDIM + nt * 16 + lr;
      O[((size_t)b * SEQ + srow) * DMODEL + col] = f2bf(oacc[nt][r] / lrun[r]);
    }
}

// -------------------------------------------------------------------- driver
extern "C" void kernel_launch(void* const* d_in, const int* in_sizes, int n_in,
                              void* d_out, int out_size, void* d_ws,
                              size_t ws_size, hipStream_t stream) {
  (void)in_sizes; (void)n_in; (void)out_size; (void)ws_size;
  const float* x  = (const float*)d_in[0];
  const float* Wq = (const float*)d_in[1];
  const float* Wk = (const float*)d_in[2];
  const float* Wv = (const float*)d_in[3];
  const float* Wo = (const float*)d_in[4];

  char* ws = (char*)d_ws;
  size_t off = 0;
  auto alloc = [&](size_t bytes) {
    char* p = ws + off;
    off += (bytes + 255) & ~(size_t)255;
    return (unsigned short*)p;
  };
  unsigned short* xb   = alloc((size_t)MROWS * DMODEL * 2);
  unsigned short* Wqb  = alloc((size_t)DMODEL * DMODEL * 2);
  unsigned short* Wkb  = alloc((size_t)KVDIM * DMODEL * 2);
  unsigned short* Wvb  = alloc((size_t)KVDIM * DMODEL * 2);
  unsigned short* Wob  = alloc((size_t)DMODEL * DMODEL * 2);
  unsigned short* Qrow = alloc((size_t)MROWS * DMODEL * 2);
  unsigned short* Krow = alloc((size_t)MROWS * KVDIM * 2);
  unsigned short* Vrow = alloc((size_t)MROWS * KVDIM * 2);
  unsigned short* Qb   = alloc((size_t)MROWS * DMODEL * 2);
  unsigned short* Kb   = alloc((size_t)MROWS * KVDIM * 2);
  unsigned short* Vb   = alloc((size_t)MROWS * KVDIM * 2);
  unsigned short* Ob   = alloc((size_t)MROWS * DMODEL * 2);

  // stage 1: f32 -> bf16
  cvt_f32_bf16<<<4096, 256, 0, stream>>>(x,  xb,  MROWS * DMODEL);
  cvt_f32_bf16<<<2048, 256, 0, stream>>>(Wq, Wqb, DMODEL * DMODEL);
  cvt_f32_bf16<<<1024, 256, 0, stream>>>(Wk, Wkb, KVDIM * DMODEL);
  cvt_f32_bf16<<<1024, 256, 0, stream>>>(Wv, Wvb, KVDIM * DMODEL);
  cvt_f32_bf16<<<2048, 256, 0, stream>>>(Wo, Wob, DMODEL * DMODEL);

  // stage 2: QKV projections (WMMA GEMM, bf16 out)
  gemm_bf16_nt<false><<<dim3(DMODEL / 128, MROWS / 128), 256, 0, stream>>>(
      xb, Wqb, Qrow, MROWS, DMODEL, DMODEL);
  gemm_bf16_nt<false><<<dim3(KVDIM / 128, MROWS / 128), 256, 0, stream>>>(
      xb, Wkb, Krow, MROWS, KVDIM, DMODEL);
  gemm_bf16_nt<false><<<dim3(KVDIM / 128, MROWS / 128), 256, 0, stream>>>(
      xb, Wvb, Vrow, MROWS, KVDIM, DMODEL);

  // stage 3: RoPE + scatter to head-major
  rope_scatter<<<4096, 256, 0, stream>>>(Qrow, Qb, NHEADS, 1);
  rope_scatter<<<1024, 256, 0, stream>>>(Krow, Kb, NKV, 1);
  rope_scatter<<<1024, 256, 0, stream>>>(Vrow, Vb, NKV, 0);

  // stage 4: causal flash attention (WMMA scores + WMMA P*V)
  flash_attn<<<dim3(SEQ / 64, NHEADS, BATCH), 128, 0, stream>>>(Qb, Kb, Vb, Ob);

  // stage 5: output projection, f32 straight into d_out
  gemm_bf16_nt<true><<<dim3(DMODEL / 128, MROWS / 128), 256, 0, stream>>>(
      Ob, Wob, d_out, MROWS, DMODEL, DMODEL);
}
